// ModifiedViTSelfAttention_90933047591123
// MI455X (gfx1250) — matrix operand order repaired and checked by reading, not compiled, not run on hardware
//
#include <hip/hip_runtime.h>
#include <hip/hip_bf16.h>

typedef __bf16 bf16;
typedef __attribute__((ext_vector_type(16))) __bf16 v16bf;
typedef __attribute__((ext_vector_type(8)))  __bf16 v8bf;
typedef __attribute__((ext_vector_type(8)))  float  v8f;
typedef int v4i __attribute__((vector_size(16)));   // int4 (builtin arg type)

constexpr int Bsz = 64, S = 577, Dm = 768, H = 12, DH = 64, KQ = 289;
constexpr int SPAD = 608;    // S padded to 19*32 for 32-wide k-steps in attention
constexpr int KQPAD = 304;   // KQ padded to 19*16 query tiles
constexpr int NIT = SPAD / 32;  // 19 attention k-steps
constexpr float NEGBIG = -3.0e38f;

// ---- CDNA5 feature detection ---------------------------------------------
#if defined(__has_builtin)
#if __has_builtin(__builtin_amdgcn_tensor_load_to_lds) && \
    __has_builtin(__builtin_amdgcn_s_wait_tensorcnt)
#define HAVE_TDM 1
#endif
#if __has_builtin(__builtin_amdgcn_global_load_async_to_lds_b128)
#define HAVE_ASYNC 1
#endif
#if __has_builtin(__builtin_amdgcn_sched_barrier)
#define SCHED_BARRIER() __builtin_amdgcn_sched_barrier(0)
#endif
#endif
#ifndef SCHED_BARRIER
#define SCHED_BARRIER()
#endif

#ifdef HAVE_ASYNC
typedef __attribute__((address_space(1))) v4i gas_v4i;
typedef __attribute__((address_space(3))) v4i las_v4i;
#define ASYNC_B128(gp, lp) \
  __builtin_amdgcn_global_load_async_to_lds_b128((gas_v4i*)(gp), (las_v4i*)(lp), 0, 0)
#if __has_builtin(__builtin_amdgcn_s_wait_asynccnt)
#define WAIT_ASYNC(n) do { __builtin_amdgcn_s_wait_asynccnt(n); asm volatile("" ::: "memory"); } while (0)
#else
#define WAIT_ASYNC(n) asm volatile("s_wait_asynccnt %0" :: "i"(n) : "memory")
#endif
#endif

#ifdef HAVE_TDM
typedef unsigned int u32;
typedef u32 v4u __attribute__((vector_size(16)));
typedef int  v8i __attribute__((vector_size(32)));

static __device__ __forceinline__ unsigned lds_off_of(void* p) {
  return (unsigned)(unsigned long long)(__attribute__((address_space(3))) void*)p;
}

// 2D bf16 tile DMA: global (row-major, stride_x elems/row) -> LDS, with the
// TDM's pad feature inserting `pad_amount` DWORDs every `pad_interval` code
// of stored DWORDs (LDS bank spreading). Descriptor per ISA cdna5 8.3/8.4.
static __device__ __forceinline__ void tdm_load_2d(
    const bf16* gaddr, unsigned lds_off,
    u32 tile_x, u32 tile_y, u32 tensor_x, u32 tensor_y,
    u32 stride_x, u32 pad_interval, u32 pad_amount) {
  const unsigned long long ga = (unsigned long long)gaddr;
  v4u g0;
  g0[0] = 1u;                                             // count=1 (valid user D#)
  g0[1] = lds_off;                                        // lds_addr (bytes)
  g0[2] = (u32)ga;                                        // global_addr[31:0]
  g0[3] = (u32)((ga >> 32) & 0x1FFFFFFull) | (2u << 30);  // ga[56:32] | type=2
  v8i g1;
  g1[0] = (int)((1u << 16)                 // data_size = 2 bytes
              | (1u << 20)                 // pad_enable
              | (pad_interval << 22)       // pad_interval code (bits 24:22)
              | (pad_amount << 25));       // pad_amount code (bits 31:25)
  g1[1] = (int)((tensor_x & 0xFFFFu) << 16);                               // dim0[15:0]
  g1[2] = (int)(((tensor_x >> 16) & 0xFFFFu) | ((tensor_y & 0xFFFFu) << 16));
  g1[3] = (int)(((tensor_y >> 16) & 0xFFFFu) | ((tile_x & 0xFFFFu) << 16)); // tile_dim0
  g1[4] = (int)(tile_y & 0xFFFFu);         // tile_dim1 (tile_dim2 = 0)
  g1[5] = (int)stride_x;                   // tensor_dim0_stride[31:0]
  g1[6] = 0;
  g1[7] = 0;
  v4i g2 = {0, 0, 0, 0};
  v4i g3 = {0, 0, 0, 0};
#if __clang_major__ >= 23
  v8i g4 = {0, 0, 0, 0, 0, 0, 0, 0};
  __builtin_amdgcn_tensor_load_to_lds(g0, g1, g2, g3, g4, 0);
#else
  __builtin_amdgcn_tensor_load_to_lds(g0, g1, g2, g3, 0);
#endif
}
#define WAIT_TENSOR(n) do { __builtin_amdgcn_s_wait_tensorcnt((unsigned short)(n)); \
                            asm volatile("" ::: "memory"); } while (0)
#endif

#if defined(HAVE_TDM) || defined(HAVE_ASYNC)
#define STAGED 1
#endif

// ---------------------------------------------------------------------------
// f32 -> bf16 convert (weights; tiny, L2-resident)
// ---------------------------------------------------------------------------
__global__ __launch_bounds__(256) void cvt_f32_to_bf16(const float* __restrict__ src,
                                                       bf16* __restrict__ dst, int n) {
  int i = (blockIdx.x * 256 + threadIdx.x) * 4;
  if (i >= n) return;
  float4 f = *(const float4*)(src + i);
  dst[i + 0] = (bf16)f.x;
  dst[i + 1] = (bf16)f.y;
  dst[i + 2] = (bf16)f.z;
  dst[i + 3] = (bf16)f.w;
}

// ---------------------------------------------------------------------------
// Fused K & V projection. out = hid @ W^T + b.
// Wave tile: 16 rows x 64 cols (one head) for BOTH K and V (A-frag shared).
// sched_barrier keeps the 8 B-fragment loads as one clause ahead of the
// 8 WMMAs so the waits drain progressively instead of one-load-one-wmma.
// K stored [b,h,s,dh] (row-major), V stored transposed [b,h,dh,s].
// ---------------------------------------------------------------------------
__global__ __launch_bounds__(256) void gemm_kv_kernel(
    const float* __restrict__ hid,
    const bf16* __restrict__ wk, const float* __restrict__ bk,
    const bf16* __restrict__ wv, const float* __restrict__ bv,
    bf16* __restrict__ Kb, bf16* __restrict__ Vt) {
  const int lane = threadIdx.x & 31;
  const int wave = threadIdx.x >> 5;
  const int l16 = lane & 15;
  const int hlf = lane >> 4;
  const int tile = blockIdx.x * 8 + wave;
  const int mt = tile / H;   // 16-row tile index (exactly B*S/16 = 2308 tiles)
  const int h  = tile % H;   // head == 64-col tile

  const v8f vzero = {0.f, 0.f, 0.f, 0.f, 0.f, 0.f, 0.f, 0.f};
  v8f acK[4], acV[4];
#pragma unroll
  for (int j = 0; j < 4; ++j) { acK[j] = vzero; acV[j] = vzero; }

  const float* arow = hid + (long)(mt * 16 + l16) * Dm;
  for (int k0 = 0; k0 < Dm; k0 += 32) {
    // issue ALL loads for this k-step first (A f32 chunks + 8 B fragments)
    v8f a1 = *(const v8f*)(arow + k0 + hlf * 16);
    v8f a2 = *(const v8f*)(arow + k0 + hlf * 16 + 8);
    v16bf bKf[4], bVf[4];
#pragma unroll
    for (int nt = 0; nt < 4; ++nt) {
      const long woff = (long)(h * 64 + nt * 16 + l16) * Dm + k0 + hlf * 16;
      bKf[nt] = *(const v16bf*)(wk + woff);
      bVf[nt] = *(const v16bf*)(wv + woff);
    }
    SCHED_BARRIER();   // loads stay above, WMMAs below
    v16bf a;
#pragma unroll
    for (int e = 0; e < 8; ++e) {
      ((bf16*)&a)[e]     = (bf16)a1[e];
      ((bf16*)&a)[e + 8] = (bf16)a2[e];
    }
#pragma unroll
    for (int nt = 0; nt < 4; ++nt) {
      acK[nt] = __builtin_amdgcn_wmma_f32_16x16x32_bf16(false, a, false, bKf[nt],
                                                        (short)0, acK[nt], false, false);
      acV[nt] = __builtin_amdgcn_wmma_f32_16x16x32_bf16(false, a, false, bVf[nt],
                                                        (short)0, acV[nt], false, false);
    }
  }
  // bias + stores. C layout: VGPR e -> row e+8*hlf, col l16.
#pragma unroll
  for (int nt = 0; nt < 4; ++nt) {
    const int dh = nt * 16 + l16;
    const int n  = h * 64 + dh;
    const float bkn = bk[n];
    const float bvn = bv[n];
#pragma unroll
    for (int e = 0; e < 8; ++e) {
      const int row = mt * 16 + e + 8 * hlf;
      const int b = row / S;
      const int s = row - b * S;
      Kb[(((long)(b * H + h)) * SPAD + s) * DH + dh] = (bf16)(acK[nt][e] + bkn);
      Vt[(((long)(b * H + h)) * DH + dh) * SPAD + s] = (bf16)(acV[nt][e] + bvn);
    }
  }
}

// ---------------------------------------------------------------------------
// Q projection with fused gather of patch_indices_to_keep.
// M = B*KQ = 18496 rows (exactly 1156 tiles). Stored [b,h,j,dh].
// ---------------------------------------------------------------------------
__global__ __launch_bounds__(256) void gemm_q_kernel(
    const float* __restrict__ hid,
    const bf16* __restrict__ wq, const float* __restrict__ bq,
    const int* __restrict__ idx,
    bf16* __restrict__ Qb) {
  const int lane = threadIdx.x & 31;
  const int wave = threadIdx.x >> 5;
  const int l16 = lane & 15;
  const int hlf = lane >> 4;
  const int tile = blockIdx.x * 8 + wave;
  const int mt = tile / H;
  const int h  = tile % H;

  const v8f vzero = {0.f, 0.f, 0.f, 0.f, 0.f, 0.f, 0.f, 0.f};
  v8f ac[4];
#pragma unroll
  for (int j = 0; j < 4; ++j) ac[j] = vzero;

  const int grow = mt * 16 + l16;
  const int b0 = grow / KQ;
  const int j0 = grow - b0 * KQ;
  const float* arow = hid + ((long)b0 * S + idx[j0]) * Dm;

  for (int k0 = 0; k0 < Dm; k0 += 32) {
    v8f a1 = *(const v8f*)(arow + k0 + hlf * 16);
    v8f a2 = *(const v8f*)(arow + k0 + hlf * 16 + 8);
    v16bf bQf[4];
#pragma unroll
    for (int nt = 0; nt < 4; ++nt) {
      const long woff = (long)(h * 64 + nt * 16 + l16) * Dm + k0 + hlf * 16;
      bQf[nt] = *(const v16bf*)(wq + woff);
    }
    SCHED_BARRIER();
    v16bf a;
#pragma unroll
    for (int e = 0; e < 8; ++e) {
      ((bf16*)&a)[e]     = (bf16)a1[e];
      ((bf16*)&a)[e + 8] = (bf16)a2[e];
    }
#pragma unroll
    for (int nt = 0; nt < 4; ++nt)
      ac[nt] = __builtin_amdgcn_wmma_f32_16x16x32_bf16(false, a, false, bQf[nt],
                                                       (short)0, ac[nt], false, false);
  }
#pragma unroll
  for (int nt = 0; nt < 4; ++nt) {
    const int dh = nt * 16 + l16;
    const float bqn = bq[h * 64 + dh];
#pragma unroll
    for (int e = 0; e < 8; ++e) {
      const int row = mt * 16 + e + 8 * hlf;
      const int b = row / KQ;
      const int j = row - b * KQ;
      Qb[(((long)(b * H + h)) * KQPAD + j) * DH + dh] = (bf16)(ac[nt][e] + bqn);
    }
  }
}

// ---------------------------------------------------------------------------
// Flash attention: one wave per (qtile, h, b). 19 k-steps of 32 columns,
// online softmax, 8 WMMAs/step. K/V tiles double-buffered into LDS by the
// Tensor Data Mover (2 DMA ops/step, TENSORcnt) or async-to-LDS fallback.
// ---------------------------------------------------------------------------
__global__ __launch_bounds__(32) void attn_kernel(
    const bf16* __restrict__ Qb, const bf16* __restrict__ Kb,
    const bf16* __restrict__ Vt, float* __restrict__ out) {
#ifdef STAGED
  __shared__ bf16 kbuf[2][32 * 72];   // K tile: 32 keys x 64 d, row stride 72
  __shared__ bf16 vbuf[2][64 * 40];   // V^T tile: 64 dh x 32 n, row stride 40
#endif
  __shared__ bf16 ptile[16 * 32];

  const int lane = threadIdx.x & 31;
  const int l16 = lane & 15;
  const int hlf = lane >> 4;
  const int qt = blockIdx.x;   // 0..18
  const int h  = blockIdx.y;
  const int b  = blockIdx.z;

  const bf16* Qp = Qb + ((long)(b * H + h) * KQPAD) * DH;
  const bf16* Kp = Kb + ((long)(b * H + h) * SPAD) * DH;
  const bf16* Vp = Vt + ((long)(b * H + h) * DH) * SPAD;

  // Q A-fragments for k-dim DH=64 -> two 16x32 fragments, loaded once
  const bf16* qrow = Qp + (long)(qt * 16 + l16) * DH;
  v16bf qa0, qa1;
  ((v8bf*)&qa0)[0] = *(const v8bf*)(qrow + hlf * 16);
  ((v8bf*)&qa0)[1] = *(const v8bf*)(qrow + hlf * 16 + 8);
  ((v8bf*)&qa1)[0] = *(const v8bf*)(qrow + 32 + hlf * 16);
  ((v8bf*)&qa1)[1] = *(const v8bf*)(qrow + 32 + hlf * 16 + 8);

  const v8f vzero = {0.f, 0.f, 0.f, 0.f, 0.f, 0.f, 0.f, 0.f};
  v8f acc[4];
#pragma unroll
  for (int j = 0; j < 4; ++j) acc[j] = vzero;
  float mrow[8], lrow[8];
#pragma unroll
  for (int e = 0; e < 8; ++e) { mrow[e] = NEGBIG; lrow[e] = 0.f; }

#if defined(HAVE_TDM)
  const unsigned kofs0 = lds_off_of(&kbuf[0][0]), kofs1 = lds_off_of(&kbuf[1][0]);
  const unsigned vofs0 = lds_off_of(&vbuf[0][0]), vofs1 = lds_off_of(&vbuf[1][0]);
  auto stage = [&](int it, int buf) {
    const int n0 = it * 32;
    // K: 32 rows x 64 elems, row stride 64; LDS pad 4 DW per 32 DW -> stride 72
    tdm_load_2d(Kp + (long)n0 * DH, buf ? kofs1 : kofs0,
                64, 32, 64, (u32)(SPAD - n0), 64, /*pi=32DW*/4, /*pa=4DW*/3);
    // V^T: 64 rows x 32 elems, row stride SPAD; LDS pad 4 DW per 16 DW -> 40
    tdm_load_2d(Vp + n0, buf ? vofs1 : vofs0,
                32, 64, (u32)(SPAD - n0), 64, SPAD, /*pi=16DW*/3, /*pa=4DW*/3);
  };
#define WAIT_STAGE_PREF() WAIT_TENSOR(2)
#define WAIT_STAGE_LAST() WAIT_TENSOR(0)
#elif defined(HAVE_ASYNC)
  auto stage = [&](int it, int buf) {
    const int n0 = it * 32;
#pragma unroll
    for (int i = 0; i < 8; ++i)   // K: 32 rows x 128B -> lane=row, 8 chunks
      ASYNC_B128(Kp + (long)(n0 + lane) * DH + i * 8, &kbuf[buf][lane * 72 + i * 8]);
#pragma unroll
    for (int i = 0; i < 8; ++i) { // V^T: 64 rows x 64B -> 256 chunks
      const int j = i * 32 + lane;
      const int dh = j >> 2, ch = j & 3;
      ASYNC_B128(Vp + (long)dh * SPAD + n0 + ch * 8, &vbuf[buf][dh * 40 + ch * 8]);
    }
  };
#define WAIT_STAGE_PREF() WAIT_ASYNC(16)
#define WAIT_STAGE_LAST() WAIT_ASYNC(0)
#endif

#ifdef STAGED
  stage(0, 0);
#endif

  for (int it = 0; it < NIT; ++it) {
    const int n0 = it * 32;
    const int c0 = n0 + l16;
    const int c1 = c0 + 16;

    v16bf kb00, kb01, kb10, kb11, vb[4];
#ifdef STAGED
    const int cur = it & 1;
    if (it + 1 < NIT) { stage(it + 1, cur ^ 1); WAIT_STAGE_PREF(); }
    else              { WAIT_STAGE_LAST(); }
    // K B-fragments from LDS: lane col = local key row, 16 contiguous d
    {
      const bf16* k0p = &kbuf[cur][l16 * 72 + hlf * 16];
      const bf16* k1p = &kbuf[cur][(16 + l16) * 72 + hlf * 16];
      kb00 = *(const v16bf*)(k0p);
      kb01 = *(const v16bf*)(k0p + 32);
      kb10 = *(const v16bf*)(k1p);
      kb11 = *(const v16bf*)(k1p + 32);
#pragma unroll
      for (int j = 0; j < 4; ++j)
        vb[j] = *(const v16bf*)(&vbuf[cur][(j * 16 + l16) * 40 + hlf * 16]);
    }
#else
    // direct global fragment loads (V hoisted: independent of softmax)
    {
      const bf16* k0p = Kp + (long)c0 * DH + hlf * 16;
      const bf16* k1p = Kp + (long)c1 * DH + hlf * 16;
      kb00 = *(const v16bf*)(k0p);
      kb01 = *(const v16bf*)(k0p + 32);
      kb10 = *(const v16bf*)(k1p);
      kb11 = *(const v16bf*)(k1p + 32);
#pragma unroll
      for (int j = 0; j < 4; ++j)
        vb[j] = *(const v16bf*)(Vp + (long)(j * 16 + l16) * SPAD + n0 + hlf * 16);
    }
#endif

    v8f s0 = __builtin_amdgcn_wmma_f32_16x16x32_bf16(false, qa0, false, kb00,
                                                     (short)0, vzero, false, false);
    s0 = __builtin_amdgcn_wmma_f32_16x16x32_bf16(false, qa1, false, kb01,
                                                 (short)0, s0, false, false);
    v8f s1 = __builtin_amdgcn_wmma_f32_16x16x32_bf16(false, qa0, false, kb10,
                                                     (short)0, vzero, false, false);
    s1 = __builtin_amdgcn_wmma_f32_16x16x32_bf16(false, qa1, false, kb11,
                                                 (short)0, s1, false, false);

    // scale (1/sqrt(64)) + column mask for s >= 577
    const bool v0 = (c0 < S), v1 = (c1 < S);
#pragma unroll
    for (int e = 0; e < 8; ++e) {
      s0[e] = v0 ? s0[e] * 0.125f : NEGBIG;
      s1[e] = v1 ? s1[e] * 0.125f : NEGBIG;
    }
    // per-row max over 32 columns (butterfly across the 16-lane half)
    float tmax[8];
#pragma unroll
    for (int e = 0; e < 8; ++e) tmax[e] = fmaxf(s0[e], s1[e]);
#pragma unroll
    for (int msk = 1; msk <= 8; msk <<= 1)
#pragma unroll
      for (int e = 0; e < 8; ++e)
        tmax[e] = fmaxf(tmax[e], __shfl_xor(tmax[e], msk, 32));

    float mn[8], corr[8], p0[8], p1[8], rs[8];
#pragma unroll
    for (int e = 0; e < 8; ++e) {
      mn[e]   = fmaxf(mrow[e], tmax[e]);
      corr[e] = __expf(mrow[e] - mn[e]);
      p0[e]   = __expf(s0[e] - mn[e]);
      p1[e]   = __expf(s1[e] - mn[e]);
      rs[e]   = p0[e] + p1[e];
    }
#pragma unroll
    for (int msk = 1; msk <= 8; msk <<= 1)
#pragma unroll
      for (int e = 0; e < 8; ++e)
        rs[e] += __shfl_xor(rs[e], msk, 32);
#pragma unroll
    for (int e = 0; e < 8; ++e) {
      lrow[e] = lrow[e] * corr[e] + rs[e];
      mrow[e] = mn[e];
    }
#pragma unroll
    for (int j = 0; j < 4; ++j)
#pragma unroll
      for (int e = 0; e < 8; ++e) acc[j][e] *= corr[e];

    // P (C layout) -> LDS row-major [16][32] -> reload as A fragment
#pragma unroll
    for (int e = 0; e < 8; ++e) {
      ptile[(e + 8 * hlf) * 32 + l16]      = (bf16)p0[e];
      ptile[(e + 8 * hlf) * 32 + 16 + l16] = (bf16)p1[e];
    }
    __syncthreads();
    v16bf pa;
    const bf16* pr = ptile + l16 * 32;
    ((v8bf*)&pa)[0] = *(const v8bf*)(pr + hlf * 16);
    ((v8bf*)&pa)[1] = *(const v8bf*)(pr + hlf * 16 + 8);
    __syncthreads();

#pragma unroll
    for (int j = 0; j < 4; ++j)
      acc[j] = __builtin_amdgcn_wmma_f32_16x16x32_bf16(false, pa, false, vb[j],
                                                       (short)0, acc[j], false, false);
  }

  // normalize + write ctx: out[b, qr, h*64 + dh], f32
#pragma unroll
  for (int e = 0; e < 8; ++e) {
    const int qr = qt * 16 + e + 8 * hlf;
    if (qr < KQ) {
      const float inv = 1.0f / lrow[e];
      float* orow = out + ((long)b * KQ + qr) * Dm + h * DH;
#pragma unroll
      for (int j = 0; j < 4; ++j) orow[j * 16 + l16] = acc[j][e] * inv;
    }
  }
}

// ---------------------------------------------------------------------------
extern "C" void kernel_launch(void* const* d_in, const int* in_sizes, int n_in,
                              void* d_out, int out_size, void* d_ws, size_t ws_size,
                              hipStream_t stream) {
  const float* hid = (const float*)d_in[0];
  const float* Wq  = (const float*)d_in[1];
  const float* bq  = (const float*)d_in[2];
  const float* Wk  = (const float*)d_in[3];
  const float* bk  = (const float*)d_in[4];
  const float* Wv  = (const float*)d_in[5];
  const float* bv  = (const float*)d_in[6];
  const int*   idx = (const int*)d_in[7];
  float* out = (float*)d_out;

  constexpr long WN = (long)Dm * Dm;                // 589824 per weight
  constexpr long QN = (long)Bsz * H * KQPAD * DH;   // Q bf16 elements
  constexpr long KN = (long)Bsz * H * SPAD * DH;    // K (and V^T) bf16 elements
  bf16* wqb = (bf16*)d_ws;
  bf16* wkb = wqb + WN;
  bf16* wvb = wkb + WN;
  bf16* Qb  = wvb + WN;
  bf16* Kb  = Qb + QN;
  bf16* Vt  = Kb + KN;

  // zero Q/K/V regions so padded rows/cols are exact zeros (no NaN into WMMA)
  (void)hipMemsetAsync(Qb, 0, (size_t)(QN + 2 * KN) * sizeof(bf16), stream);

  cvt_f32_to_bf16<<<(int)(WN / 1024), 256, 0, stream>>>(Wq, wqb, (int)WN);
  cvt_f32_to_bf16<<<(int)(WN / 1024), 256, 0, stream>>>(Wk, wkb, (int)WN);
  cvt_f32_to_bf16<<<(int)(WN / 1024), 256, 0, stream>>>(Wv, wvb, (int)WN);

  // (B*S/16) m-tiles x H col-tiles, 8 waves/block -> 3462 blocks
  gemm_kv_kernel<<<(Bsz * S / 16) * H / 8, 256, 0, stream>>>(hid, wkb, bk, wvb, bv, Kb, Vt);
  // (B*KQ/16) m-tiles x H col-tiles -> 1734 blocks
  gemm_q_kernel<<<(Bsz * KQ / 16) * H / 8, 256, 0, stream>>>(hid, wqb, bq, idx, Qb);

  dim3 grid(KQPAD / 16, H, Bsz);
  attn_kernel<<<grid, 32, 0, stream>>>(Qb, Kb, Vt, out);
}
